// LSRNNBlock_69114613727829
// MI455X (gfx1250) — matrix operand order repaired and checked by reading, not compiled, not run on hardware
//
#include <hip/hip_runtime.h>
#include <hip/hip_bf16.h>

// Problem constants (fixed by the reference).
#define Bz 16
#define Lz 1024
#define Dz 512
#define Hz 256
#define Kz 4

typedef __bf16 bf16_t;
typedef __attribute__((ext_vector_type(16))) __bf16 v16bf;
typedef __attribute__((ext_vector_type(8)))  __bf16 v8bf;
typedef __attribute__((ext_vector_type(8)))  float  v8f;

// Fast transcendental helpers -> v_log_f32 / v_exp_f32 (TRANS32 ops).
__device__ __forceinline__ float flog2(float v) { return __builtin_amdgcn_logf(v); }
__device__ __forceinline__ float fexp2(float v) { return __builtin_amdgcn_exp2f(v); }
__device__ __forceinline__ float fexp(float v)  { return fexp2(v * 1.4426950408889634f); }

// ---------------------------------------------------------------------------
// Kernel 1: split-bf16 cast of x (B*L*D) and W_B (H*D):  v = hi + lo
// ---------------------------------------------------------------------------
__global__ void cast_split_kernel(const float* __restrict__ x,
                                  const float* __restrict__ WB,
                                  __hip_bfloat16* __restrict__ xhi,
                                  __hip_bfloat16* __restrict__ xlo,
                                  __hip_bfloat16* __restrict__ whi,
                                  __hip_bfloat16* __restrict__ wlo) {
  const size_t NX = (size_t)Bz * Lz * Dz;
  const size_t NW = (size_t)Hz * Dz;
  size_t idx = (size_t)blockIdx.x * blockDim.x + threadIdx.x;
  const size_t stride = (size_t)gridDim.x * blockDim.x;
  for (; idx < NX + NW; idx += stride) {
    float v;
    __hip_bfloat16 *phi, *plo;
    size_t o;
    if (idx < NX) { v = x[idx];        phi = xhi; plo = xlo; o = idx; }
    else          { v = WB[idx - NX];  phi = whi; plo = wlo; o = idx - NX; }
    __hip_bfloat16 hi = __float2bfloat16(v);
    float rem = v - __bfloat162float(hi);
    phi[o] = hi;
    plo[o] = __float2bfloat16(rem);
  }
}

// ---------------------------------------------------------------------------
// Kernel 2: transpose A_dict[i][j][k] (float4 per (i,j)) -> dictJ[j][i][k]
// so the scan matvec reads are lane-coalesced over i.
// ---------------------------------------------------------------------------
__global__ void tdict_kernel(const float* __restrict__ dict,
                             float* __restrict__ dictJ) {
  const int tid = blockIdx.x * blockDim.x + threadIdx.x;  // 65536 (i,j) pairs
  const int i = tid >> 8;
  const int j = tid & 255;
  const float4* s = (const float4*)dict;
  float4* d = (float4*)dictJ;
  d[(size_t)j * Hz + i] = s[(size_t)i * Hz + j];
}

// ---------------------------------------------------------------------------
// Kernel 3: sel = softmax(x @ W_sel^T + b_sel)   (K = 4)
// One wave per row; lanes stride over D, wave-shuffle reduction.
// ---------------------------------------------------------------------------
__global__ __launch_bounds__(256)
void sel_softmax_kernel(const float* __restrict__ x,
                        const float* __restrict__ Wsel,
                        const float* __restrict__ bsel,
                        float* __restrict__ sel) {
  const int lane = threadIdx.x & 31;
  const int wave = threadIdx.x >> 5;
  const int row  = blockIdx.x * 8 + wave;     // 16384 rows / 8 waves per block
  const float* xr = x + (size_t)row * Dz;
  float p0 = 0.f, p1 = 0.f, p2 = 0.f, p3 = 0.f;
  for (int d = lane; d < Dz; d += 32) {
    const float xv = xr[d];
    p0 += xv * Wsel[0 * Dz + d];
    p1 += xv * Wsel[1 * Dz + d];
    p2 += xv * Wsel[2 * Dz + d];
    p3 += xv * Wsel[3 * Dz + d];
  }
  for (int off = 16; off; off >>= 1) {
    p0 += __shfl_xor(p0, off, 32);
    p1 += __shfl_xor(p1, off, 32);
    p2 += __shfl_xor(p2, off, 32);
    p3 += __shfl_xor(p3, off, 32);
  }
  if (lane == 0) {
    const float l0 = p0 + bsel[0], l1 = p1 + bsel[1];
    const float l2 = p2 + bsel[2], l3 = p3 + bsel[3];
    const float m  = fmaxf(fmaxf(l0, l1), fmaxf(l2, l3));
    const float e0 = fexp(l0 - m), e1 = fexp(l1 - m);
    const float e2 = fexp(l2 - m), e3 = fexp(l3 - m);
    const float inv = 1.f / (e0 + e1 + e2 + e3);
    float* sp = sel + (size_t)row * Kz;
    sp[0] = e0 * inv; sp[1] = e1 * inv; sp[2] = e2 * inv; sp[3] = e3 * inv;
  }
}

// ---------------------------------------------------------------------------
// Kernel 4: u = x @ W_B^T  via split-bf16 WMMA (3 wmma per 16x16x32 step).
// One 16x16 output tile per wave; M=16384, N=256, K=512.
// A fragment (16x32 bf16): lane m=lane&15, half=lane>>4 holds
//   K = k0+half*16 .. +15 (elems 0-15).
// B fragment (32x16 bf16): lane n=lane&15, half holds K = k0+half*16 .. +15,
//   contiguous in W_B row n (since B[k][n] = W_B[n][k]).
// ---------------------------------------------------------------------------
__global__ __launch_bounds__(256)
void gemm_u_kernel(const bf16_t* __restrict__ xhi, const bf16_t* __restrict__ xlo,
                   const bf16_t* __restrict__ whi, const bf16_t* __restrict__ wlo,
                   float* __restrict__ u) {
  const int lane = threadIdx.x & 31;
  const int wave = threadIdx.x >> 5;
  const int n0   = blockIdx.x * 16;
  const int m0   = (blockIdx.y * 8 + wave) * 16;
  const int half = lane >> 4;
  const size_t arow = (size_t)(m0 + (lane & 15)) * Dz;
  const size_t brow = (size_t)(n0 + (lane & 15)) * Dz;

  v8f acc = {};
  for (int k0 = 0; k0 < Dz; k0 += 32) {
    union { v16bf v; v8bf h[2]; } ahi, alo;
    const int ka = k0 + half * 16;
    ahi.h[0] = *(const v8bf*)(xhi + arow + ka);
    ahi.h[1] = *(const v8bf*)(xhi + arow + ka + 8);
    alo.h[0] = *(const v8bf*)(xlo + arow + ka);
    alo.h[1] = *(const v8bf*)(xlo + arow + ka + 8);
    const int kb = k0 + half * 16;
    const v16bf bhi = *(const v16bf*)(whi + brow + kb);
    const v16bf blo = *(const v16bf*)(wlo + brow + kb);
    acc = __builtin_amdgcn_wmma_f32_16x16x32_bf16(false, ahi.v, false, bhi,
                                                  (short)0, acc, false, false);
    acc = __builtin_amdgcn_wmma_f32_16x16x32_bf16(false, alo.v, false, bhi,
                                                  (short)0, acc, false, false);
    acc = __builtin_amdgcn_wmma_f32_16x16x32_bf16(false, ahi.v, false, blo,
                                                  (short)0, acc, false, false);
  }
  // C/D layout: VGPR r -> M = m0 + r + half*8, N = n0 + (lane&15)
  const int mbase = m0 + half * 8;
  const int n = n0 + (lane & 15);
#pragma unroll
  for (int r = 0; r < 8; ++r)
    u[(size_t)(mbase + r) * Hz + n] = acc[r];
}

// ---------------------------------------------------------------------------
// Kernel 5: precompute normInv[b,t,j] = (sum_i |sum_k s_k A[i,j,k]|^1.2)^(-1/1.2)
// Fully parallel (does not depend on h). 16 timesteps per block amortize the
// dict reads; lane-consecutive j => float4-coalesced loads of A_dict[i][j][0..3].
// ---------------------------------------------------------------------------
__global__ __launch_bounds__(256)
void norm_kernel(const float* __restrict__ dict,
                 const float* __restrict__ sel,
                 float* __restrict__ normInv) {
  const int bt0 = blockIdx.x * 16;      // 16 consecutive (b*L+t) rows
  __shared__ float4 s_sh[16];
  if (threadIdx.x < 16) {
    const float* sp = sel + (size_t)(bt0 + threadIdx.x) * Kz;
    s_sh[threadIdx.x] = make_float4(sp[0], sp[1], sp[2], sp[3]);
  }
  __syncthreads();
  const int j = threadIdx.x;
  float acc[16];
#pragma unroll
  for (int tt = 0; tt < 16; ++tt) acc[tt] = 0.f;
  const float4* d4 = (const float4*)dict;
  for (int i = 0; i < Hz; ++i) {
    const float4 d = d4[(size_t)i * Hz + j];
#pragma unroll
    for (int tt = 0; tt < 16; ++tt) {
      const float4 s = s_sh[tt];   // LDS broadcast
      const float a = fabsf(s.x * d.x + s.y * d.y + s.z * d.z + s.w * d.w);
      acc[tt] += (a > 0.f) ? fexp2(1.2f * flog2(a)) : 0.f;
    }
  }
#pragma unroll
  for (int tt = 0; tt < 16; ++tt) {
    const float sum = acc[tt];
    normInv[(size_t)(bt0 + tt) * Hz + j] =
        (sum > 0.f) ? fexp2((-1.0f / 1.2f) * flog2(sum)) : 0.f;
  }
}

// ---------------------------------------------------------------------------
// Kernel 6: sequential scan. One workgroup (32 wave32) per batch.
//   hs_j   = h_j * normInv[b,t,j]
//   h_i'   = sum_k s_k sum_j A_dict[i,j,k] * hs_j + u[b,t,i]
// dictJ is j-major: dictJ[(j*H + i)*K + k] -> lane-coalesced float4 over i.
// 1024 threads = 256 i-rows x 4 j-partitions; partials reduced through LDS.
// ---------------------------------------------------------------------------
__global__ __launch_bounds__(1024)
void scan_kernel(const float* __restrict__ dictJ,
                 const float* __restrict__ sel,
                 const float* __restrict__ normInv,
                 const float* __restrict__ u,
                 float* __restrict__ out) {
  const int b   = blockIdx.x;
  const int tid = threadIdx.x;
  __shared__ float h_sh[Hz];
  __shared__ float hs[Hz];
  __shared__ float red[4 * Hz];
  __shared__ float s_sh[4];
  if (tid < Hz) h_sh[tid] = 0.f;
  const int jp = tid >> 8;       // j partition: 0..3 (64 columns each)
  const int i  = tid & 255;      // output row
  const float4* dJ4 = (const float4*)dictJ;

  for (int t = 0; t < Lz; ++t) {
    const size_t row = (size_t)b * Lz + t;
    __syncthreads();                       // prev-step h_sh ready, red consumed
    if (tid < 4) s_sh[tid] = sel[row * Kz + tid];
    if (tid < Hz) {
      hs[tid] = h_sh[tid] * normInv[row * Hz + tid];
      if (t + 1 < Lz) {                    // warm next step's streams (global_prefetch_b8)
        __builtin_prefetch(normInv + (row + 1) * Hz + tid, 0, 1);
        __builtin_prefetch(u + (row + 1) * Hz + tid, 0, 1);
      }
    }
    __syncthreads();
    const float s0 = s_sh[0], s1 = s_sh[1], s2 = s_sh[2], s3 = s_sh[3];
    float acc = 0.f;
#pragma unroll 4
    for (int jj = 0; jj < 64; ++jj) {
      const int jcol = (jp << 6) + jj;
      const float4 d = dJ4[(size_t)jcol * Hz + i];
      acc += (s0 * d.x + s1 * d.y + s2 * d.z + s3 * d.w) * hs[jcol];
    }
    red[(jp << 8) + i] = acc;
    __syncthreads();
    if (tid < Hz) {
      const float y = red[tid] + red[Hz + tid] + red[2 * Hz + tid] +
                      red[3 * Hz + tid] + u[row * Hz + tid];
      h_sh[tid] = y;
      out[row * Hz + tid] = y;
    }
  }
}

// ---------------------------------------------------------------------------
extern "C" void kernel_launch(void* const* d_in, const int* in_sizes, int n_in,
                              void* d_out, int out_size, void* d_ws, size_t ws_size,
                              hipStream_t stream) {
  (void)in_sizes; (void)n_in; (void)out_size; (void)ws_size;
  const float* x     = (const float*)d_in[0];   // (B, L, D)
  const float* Wsel  = (const float*)d_in[1];   // (K, D)
  const float* bsel  = (const float*)d_in[2];   // (K,)
  const float* WB    = (const float*)d_in[3];   // (H, D)
  const float* Adict = (const float*)d_in[4];   // (H, H, K)
  float* out = (float*)d_out;                   // (B, L, H)

  const size_t NX = (size_t)Bz * Lz * Dz;       // 8,388,608
  const size_t NW = (size_t)Hz * Dz;            // 131,072
  const size_t NR = (size_t)Bz * Lz;            // 16,384 rows

  char* ws = (char*)d_ws;
  __hip_bfloat16* xhi = (__hip_bfloat16*)ws;  ws += NX * 2;
  __hip_bfloat16* xlo = (__hip_bfloat16*)ws;  ws += NX * 2;
  __hip_bfloat16* whi = (__hip_bfloat16*)ws;  ws += NW * 2;
  __hip_bfloat16* wlo = (__hip_bfloat16*)ws;  ws += NW * 2;
  float* sel   = (float*)ws;                  ws += NR * Kz * 4;
  float* nrmI  = (float*)ws;                  ws += NR * Hz * 4;
  float* uu    = (float*)ws;                  ws += NR * Hz * 4;
  float* dJ    = (float*)ws;                  ws += (size_t)Hz * Hz * Kz * 4;

  hipLaunchKernelGGL(cast_split_kernel, dim3(2048), dim3(256), 0, stream,
                     x, WB, xhi, xlo, whi, wlo);
  hipLaunchKernelGGL(tdict_kernel, dim3(256), dim3(256), 0, stream, Adict, dJ);
  hipLaunchKernelGGL(sel_softmax_kernel, dim3(2048), dim3(256), 0, stream,
                     x, Wsel, bsel, sel);
  hipLaunchKernelGGL(gemm_u_kernel, dim3(16, 128), dim3(256), 0, stream,
                     (const bf16_t*)xhi, (const bf16_t*)xlo,
                     (const bf16_t*)whi, (const bf16_t*)wlo, uu);
  hipLaunchKernelGGL(norm_kernel, dim3(1024), dim3(256), 0, stream,
                     Adict, sel, nrmI);
  hipLaunchKernelGGL(scan_kernel, dim3(Bz), dim3(1024), 0, stream,
                     dJ, sel, nrmI, uu, out);
}
